// MultiHeadAttention_34978213658719
// MI455X (gfx1250) — matrix operand order repaired and checked by reading, not compiled, not run on hardware
//
#include <hip/hip_runtime.h>

// MI455X / gfx1250, wave32. All matrix math via v_wmma_f32_16x16x32_bf16.
// bf16 operands materialized once; inner loops = 16B loads + WMMA.
// Attention: 32-row Q tile per wave (2 row-parts share all K/V fragments),
// global_prefetch_b8 on the next key block.

typedef __attribute__((ext_vector_type(16))) __bf16 v16bf;
typedef __attribute__((ext_vector_type(8)))  __bf16 v8bf;
typedef __attribute__((ext_vector_type(8)))  float  v8f;

#define B_  4
#define S_  2048
#define D_  1024
#define H_  16
#define DK_ 64
#define M_  (B_ * S_)   // 8192

static __device__ __forceinline__ __bf16 f2bf(float f) {
  unsigned u = __builtin_bit_cast(unsigned, f);
  u += 0x7FFFu + ((u >> 16) & 1u);            // round-to-nearest-even
  unsigned short s = (unsigned short)(u >> 16);
  return __builtin_bit_cast(__bf16, s);
}

static __device__ __forceinline__ v16bf cat8(v8bf lo, v8bf hi) {
  return __builtin_shufflevector(lo, hi, 0, 1, 2, 3, 4, 5, 6, 7,
                                 8, 9, 10, 11, 12, 13, 14, 15);
}

// ---------------------------------------------------------------------------
// One-shot f32 -> bf16 conversion (bandwidth-bound pre-pass).
// ---------------------------------------------------------------------------
__global__ __launch_bounds__(256)
void cvt_f32_to_bf16(const float* __restrict__ in, __bf16* __restrict__ out, int n)
{
  int i = (blockIdx.x * 256 + threadIdx.x) * 8;
  if (i >= n) return;
  float4 a = *(const float4*)(in + i);
  float4 b = *(const float4*)(in + i + 4);
  v8bf o;
  o[0] = f2bf(a.x); o[1] = f2bf(a.y); o[2] = f2bf(a.z); o[3] = f2bf(a.w);
  o[4] = f2bf(b.x); o[5] = f2bf(b.y); o[6] = f2bf(b.z); o[7] = f2bf(b.w);
  *(v8bf*)(out + i) = o;
}

// ---------------------------------------------------------------------------
// GEMM: Y = X @ W^T + bias.  X:[M,K] bf16, W:[N,K] bf16, bias f32.
// One wave computes a 32x64 strip: 2 A-fragments x 4 B-tiles = 8 WMMAs/K-step.
// mode 0: f32 out, plain [M,N]
// mode 1: bf16 out, head-split [B,H,S,DK]
// mode 2: bf16 out, head-split transposed [B,H,DK,S]   (for V)
// ---------------------------------------------------------------------------
__global__ __launch_bounds__(128)
void mha_gemm_bf16(const __bf16* __restrict__ X, const __bf16* __restrict__ W,
                   const float* __restrict__ bias, void* __restrict__ Y,
                   int M, int N, int K, int mode)
{
  const int lane = threadIdx.x & 31;
  const int wave = threadIdx.x >> 5;
  const int half = lane >> 4;
  const int l15  = lane & 15;

  const int mBase = (blockIdx.x * 4 + wave) * 32;   // 32 rows
  const int nBase = blockIdx.y * 64;                // 64 cols

  v8f acc[2][4];
#pragma unroll
  for (int p = 0; p < 2; ++p)
#pragma unroll
    for (int t = 0; t < 4; ++t) acc[p][t] = (v8f){};

  const __bf16* xr0 = X + (size_t)(mBase + l15) * K;
  const __bf16* xr1 = X + (size_t)(mBase + 16 + l15) * K;

  for (int kb = 0; kb < K; kb += 32) {
    // A fragment: element j -> K = kb + j + (j>=8?8:0) + 8*half
    // -> two contiguous 16B runs at kb+8*half and kb+16+8*half.
    v16bf a0 = cat8(*(const v8bf*)(xr0 + kb + half * 8),
                    *(const v8bf*)(xr0 + kb + 16 + half * 8));
    v16bf a1 = cat8(*(const v8bf*)(xr1 + kb + half * 8),
                    *(const v8bf*)(xr1 + kb + 16 + half * 8));
#pragma unroll
    for (int t = 0; t < 4; ++t) {
      // B fragment: element j -> K = kb + j + (j>=8?8:0) + 8*half; B[k][n] = W[n][kb+k].
      const __bf16* wr = W + (size_t)(nBase + t * 16 + l15) * K + kb + half * 8;
      v16bf bfr = cat8(*(const v8bf*)(wr), *(const v8bf*)(wr + 16));
      acc[0][t] = __builtin_amdgcn_wmma_f32_16x16x32_bf16(false, a0, false, bfr,
                                                          (short)0, acc[0][t],
                                                          false, false);
      acc[1][t] = __builtin_amdgcn_wmma_f32_16x16x32_bf16(false, a1, false, bfr,
                                                          (short)0, acc[1][t],
                                                          false, false);
    }
  }

  // Epilogue. C/D layout: row = e + 8*half, col = l15 (per 16-wide tile).
#pragma unroll
  for (int t = 0; t < 4; ++t) {
    int n = nBase + t * 16 + l15;
    float bv = bias[n];
#pragma unroll
    for (int p = 0; p < 2; ++p) {
#pragma unroll
      for (int e = 0; e < 8; ++e) {
        int m = mBase + p * 16 + e + half * 8;
        float v = acc[p][t][e] + bv;
        if (mode == 0) {
          ((float*)Y)[(size_t)m * N + n] = v;
        } else {
          int bb = m / S_, s = m % S_;
          int hh = n / DK_, dk = n % DK_;
          size_t idx = (mode == 1)
              ? ((((size_t)(bb * H_ + hh)) * S_ + s) * DK_ + dk)    // [B,H,S,DK]
              : ((((size_t)(bb * H_ + hh)) * DK_ + dk) * S_ + s);   // [B,H,DK,S]
          ((__bf16*)Y)[idx] = f2bf(v);
        }
      }
    }
  }
}

// ---------------------------------------------------------------------------
// Flash attention. One wave owns a 32-row Q tile (2 row-parts) of one (b,h);
// streams key blocks of 32 with online softmax. Every K and V fragment is
// reused by both row-parts (16 WMMAs per key block). Q/K: bf16 [B,H,S,DK];
// V: bf16 [B,H,DK,S] (transposed -> contiguous B-fragments). ctx: bf16 [B,S,D].
// ---------------------------------------------------------------------------
__global__ __launch_bounds__(128)
void mha_flash_attn(const __bf16* __restrict__ Qp, const __bf16* __restrict__ Kp,
                    const __bf16* __restrict__ Vt, __bf16* __restrict__ ctx)
{
  __shared__ __bf16 ldsP[4][2 * 16 * 32];   // per-wave 2KB: two 16x32 P tiles

  const int lane = threadIdx.x & 31;
  const int wave = threadIdx.x >> 5;
  const int half = lane >> 4;
  const int l15  = lane & 15;

  const int wg    = blockIdx.x * 4 + wave;       // 0 .. B*H*(S/32)-1
  const int qT    = wg & 63;                     // S/32 = 64
  const int h     = (wg >> 6) & (H_ - 1);
  const int b     = wg >> 10;
  const int qbase = qT * 32;

  const __bf16* Qh = Qp + (((size_t)b * H_ + h) * S_) * DK_;
  const __bf16* Kh = Kp + (((size_t)b * H_ + h) * S_) * DK_;
  const __bf16* Vh = Vt + (((size_t)b * H_ + h) * DK_) * S_;
  __bf16* myLds = ldsP[wave];

  // Q A-fragments: 2 row-parts x 2 d-chunks.
  v16bf qa[2][2];
#pragma unroll
  for (int p = 0; p < 2; ++p) {
    const __bf16* qrow = Qh + (size_t)(qbase + p * 16 + l15) * DK_;
    qa[p][0] = cat8(*(const v8bf*)(qrow + half * 8),
                    *(const v8bf*)(qrow + 16 + half * 8));
    qa[p][1] = cat8(*(const v8bf*)(qrow + 32 + half * 8),
                    *(const v8bf*)(qrow + 48 + half * 8));
  }

  const float scale = 0.125f;                    // 1/sqrt(DK)
  float mrow[2][8], lrow[2][8];
#pragma unroll
  for (int p = 0; p < 2; ++p)
#pragma unroll
    for (int e = 0; e < 8; ++e) { mrow[p][e] = -1e30f; lrow[p][e] = 0.0f; }
  v8f o[2][4];
#pragma unroll
  for (int p = 0; p < 2; ++p)
#pragma unroll
    for (int t = 0; t < 4; ++t) o[p][t] = (v8f){};

  for (int kb = 0; kb < qbase + 32; kb += 32) {
    // ---- prefetch next key block (K rows + V columns); gfx1250 global_prefetch
    if (kb + 32 < qbase + 32) {
      __builtin_prefetch(Kh + (size_t)(kb + 32 + lane) * DK_, 0, 1);
      __builtin_prefetch(Vh + (size_t)lane * S_ + kb + 32, 0, 1);
      __builtin_prefetch(Vh + (size_t)(32 + lane) * S_ + kb + 32, 0, 1);
    }

    // ---- K fragments: 2 key tiles x 2 d-chunks (shared by both row-parts)
    v16bf kf[2][2];
#pragma unroll
    for (int t = 0; t < 2; ++t)
#pragma unroll
      for (int c = 0; c < 2; ++c) {
        const __bf16* kr = Kh + (size_t)(kb + t * 16 + l15) * DK_ + c * 32 + half * 8;
        kf[t][c] = cat8(*(const v8bf*)(kr), *(const v8bf*)(kr + 16));
      }

    // ---- S = Q @ K^T : 4 tiles (2 parts x 2 key tiles), 2 WMMAs each
    v8f sc[2][2];
#pragma unroll
    for (int p = 0; p < 2; ++p)
#pragma unroll
      for (int t = 0; t < 2; ++t) {
        v8f s = {};
        s = __builtin_amdgcn_wmma_f32_16x16x32_bf16(false, qa[p][0], false, kf[t][0],
                                                    (short)0, s, false, false);
        s = __builtin_amdgcn_wmma_f32_16x16x32_bf16(false, qa[p][1], false, kf[t][1],
                                                    (short)0, s, false, false);
        sc[p][t] = s;
      }

    // ---- per-part: scale + causal mask, online softmax, P tile to LDS
#pragma unroll
    for (int p = 0; p < 2; ++p) {
#pragma unroll
      for (int e = 0; e < 8; ++e) {
        int q = qbase + p * 16 + e + half * 8;
        sc[p][0][e] = (kb + l15      > q) ? -1e30f : sc[p][0][e] * scale;
        sc[p][1][e] = (kb + 16 + l15 > q) ? -1e30f : sc[p][1][e] * scale;
      }
      float mnew[8], alpha[8];
#pragma unroll
      for (int e = 0; e < 8; ++e) {
        float v = fmaxf(sc[p][0][e], sc[p][1][e]);
#pragma unroll
        for (int d = 1; d < 16; d <<= 1) v = fmaxf(v, __shfl_xor(v, d, 32));
        mnew[e]     = fmaxf(mrow[p][e], v);
        alpha[e]    = __expf(mrow[p][e] - mnew[e]);
        mrow[p][e]  = mnew[e];
      }
      __bf16* pl = myLds + p * (16 * 32);
#pragma unroll
      for (int e = 0; e < 8; ++e) {
        float p0 = __expf(sc[p][0][e] - mnew[e]);
        float p1 = __expf(sc[p][1][e] - mnew[e]);
        int row = e + half * 8;
        pl[row * 32 + l15]      = f2bf(p0);
        pl[row * 32 + 16 + l15] = f2bf(p1);
        float rs = p0 + p1;
#pragma unroll
        for (int d = 1; d < 16; d <<= 1) rs += __shfl_xor(rs, d, 32);
        lrow[p][e] = lrow[p][e] * alpha[e] + rs;
        o[p][0][e] *= alpha[e]; o[p][1][e] *= alpha[e];
        o[p][2][e] *= alpha[e]; o[p][3][e] *= alpha[e];
      }
    }

    // ---- P back in A-fragment layout (same-wave DS is in-order)
    v16bf pa0 = cat8(*(const v8bf*)(myLds + l15 * 32 + half * 8),
                     *(const v8bf*)(myLds + l15 * 32 + 16 + half * 8));
    v16bf pa1 = cat8(*(const v8bf*)(myLds + 512 + l15 * 32 + half * 8),
                     *(const v8bf*)(myLds + 512 + l15 * 32 + 16 + half * 8));

    // ---- O += P @ V : each V fragment feeds both row-parts
#pragma unroll
    for (int t2 = 0; t2 < 4; ++t2) {
      const __bf16* vr = Vh + (size_t)(t2 * 16 + l15) * S_ + kb + half * 8;
      v16bf vb = cat8(*(const v8bf*)(vr), *(const v8bf*)(vr + 16));
      o[0][t2] = __builtin_amdgcn_wmma_f32_16x16x32_bf16(false, pa0, false, vb,
                                                         (short)0, o[0][t2],
                                                         false, false);
      o[1][t2] = __builtin_amdgcn_wmma_f32_16x16x32_bf16(false, pa1, false, vb,
                                                         (short)0, o[1][t2],
                                                         false, false);
    }
  }

  // ---- normalize, write merged-head bf16 context [B,S,D]
#pragma unroll
  for (int p = 0; p < 2; ++p)
#pragma unroll
    for (int e = 0; e < 8; ++e) {
      int s = qbase + p * 16 + e + half * 8;
      float inv = 1.0f / lrow[p][e];
      __bf16* dst = ctx + ((size_t)b * S_ + s) * D_ + h * DK_;
      dst[0 * 16 + l15] = f2bf(o[p][0][e] * inv);
      dst[1 * 16 + l15] = f2bf(o[p][1][e] * inv);
      dst[2 * 16 + l15] = f2bf(o[p][2][e] * inv);
      dst[3 * 16 + l15] = f2bf(o[p][3][e] * inv);
    }
}

// ---------------------------------------------------------------------------
extern "C" void kernel_launch(void* const* d_in, const int* in_sizes, int n_in,
                              void* d_out, int out_size, void* d_ws, size_t ws_size,
                              hipStream_t stream) {
  (void)in_sizes; (void)n_in; (void)out_size; (void)ws_size;
  const float* q   = (const float*)d_in[0];
  const float* k   = (const float*)d_in[1];
  const float* v   = (const float*)d_in[2];
  /* d_in[3] = tril mask — applied analytically */
  const float* W_q = (const float*)d_in[4];
  const float* b_q = (const float*)d_in[5];
  const float* W_k = (const float*)d_in[6];
  const float* b_k = (const float*)d_in[7];
  const float* W_v = (const float*)d_in[8];
  const float* b_v = (const float*)d_in[9];
  const float* W_o = (const float*)d_in[10];
  const float* b_o = (const float*)d_in[11];
  float* out = (float*)d_out;

  const size_t act = (size_t)M_ * D_;        // 8.39M elems
  const size_t wel = (size_t)D_ * D_;        // 1.05M elems
  __bf16* Xq  = (__bf16*)d_ws;
  __bf16* Xk  = Xq  + act;
  __bf16* Xv  = Xk  + act;
  __bf16* Wqb = Xv  + act;
  __bf16* Wkb = Wqb + wel;
  __bf16* Wvb = Wkb + wel;
  __bf16* Wob = Wvb + wel;
  __bf16* Qp  = Wob + wel;
  __bf16* Kp  = Qp  + act;
  __bf16* Vt  = Kp  + act;
  __bf16* ctx = Vt  + act;

  // --- pre-pass: one-time f32 -> bf16 conversions
  int ab = (int)(act / (256 * 8));
  int wb = (int)(wel / (256 * 8));
  cvt_f32_to_bf16<<<ab, 256, 0, stream>>>(q,   Xq,  (int)act);
  cvt_f32_to_bf16<<<ab, 256, 0, stream>>>(k,   Xk,  (int)act);
  cvt_f32_to_bf16<<<ab, 256, 0, stream>>>(v,   Xv,  (int)act);
  cvt_f32_to_bf16<<<wb, 256, 0, stream>>>(W_q, Wqb, (int)wel);
  cvt_f32_to_bf16<<<wb, 256, 0, stream>>>(W_k, Wkb, (int)wel);
  cvt_f32_to_bf16<<<wb, 256, 0, stream>>>(W_v, Wvb, (int)wel);
  cvt_f32_to_bf16<<<wb, 256, 0, stream>>>(W_o, Wob, (int)wel);

  // --- projections (32x64 strip per wave): grid (M/128, N/64)
  dim3 ggrid(M_ / 128, D_ / 64);
  mha_gemm_bf16<<<ggrid, 128, 0, stream>>>(Xq, Wqb, b_q, Qp, M_, D_, D_, 1);
  mha_gemm_bf16<<<ggrid, 128, 0, stream>>>(Xk, Wkb, b_k, Kp, M_, D_, D_, 1);
  mha_gemm_bf16<<<ggrid, 128, 0, stream>>>(Xv, Wvb, b_v, Vt, M_, D_, D_, 2);

  // --- attention: B*H*(S/32) waves, 4 per block
  int nwaves = B_ * H_ * (S_ / 32);
  mha_flash_attn<<<nwaves / 4, 128, 0, stream>>>(Qp, Kp, Vt, ctx);

  // --- output projection -> f32 [B,S,D]
  mha_gemm_bf16<<<ggrid, 128, 0, stream>>>(ctx, Wob, b_o, out, M_, D_, D_, 0);
}
